// CharRNN_21406117003946
// MI455X (gfx1250) — matrix operand order repaired and checked by reading, not compiled, not run on hardware
//
#include <hip/hip_runtime.h>

#define VOCAB  96
#define EMBED  32
#define HIDDEN 128
#define BATCH  512
#define TSTEPS 2048
#define TCH    32          // timesteps of indices staged per chunk
#define MROWS  16          // batch rows per workgroup (WMMA M)

typedef __bf16 bf16_t;
typedef __attribute__((ext_vector_type(16))) bf16_t       v16bf;
typedef __attribute__((ext_vector_type(8)))  float        v8f;
typedef __attribute__((ext_vector_type(8)))  unsigned int v8u;

// f32 -> bf16 with round-to-nearest-even
static __device__ __forceinline__ unsigned short f2bf(float f) {
    union { float f; unsigned u; } v; v.f = f;
    unsigned r = v.u + 0x7fffu + ((v.u >> 16) & 1u);
    return (unsigned short)(r >> 16);
}

// CDNA5 hardware transcendental tanh (TRANS32 pipe). The trailing s_delay_alu
// covers the ISA's "1 op after TRANS before result use" hazard, since the
// compiler cannot see through the asm.
static __device__ __forceinline__ float htanh(float x) {
    float r;
    asm("v_tanh_f32 %0, %1\n\t"
        "s_delay_alu instid0(TRANS32_DEP_1)"
        : "=v"(r) : "v"(x));
    return r;
}

// Load a 16x32 bf16 A-matrix fragment from LDS. ISA layout: lane m = lane&15;
// lanes<16 hold K 0..7 & 16..23, lanes>=16 hold K 8..15 & 24..31 (2 bf16/VGPR).
// u4idx addresses uint4 (8 bf16) units.
static __device__ __forceinline__ v16bf load_lds_afrag(const unsigned short* base,
                                                       int u4idx) {
    const uint4* p = (const uint4*)base;
    uint4 lo = p[u4idx];       // K = klo+0..7
    uint4 hi = p[u4idx + 2];   // K = klo+16..23
    v8u r;
    r[0] = lo.x; r[1] = lo.y; r[2] = lo.z; r[3] = lo.w;
    r[4] = hi.x; r[5] = hi.y; r[6] = hi.z; r[7] = hi.w;
    return __builtin_bit_cast(v16bf, r);
}

// Build a 32x16 bf16 B-matrix fragment from a global f32 weight row (row = the
// N column of W, since B = W^T). ISA layout: lane n = lane&15; lanes<16 hold
// K 0..7 & 16..23, lanes>=16 hold K 8..15 & 24..31 (kb = 16*half).
static __device__ __forceinline__ v16bf load_w_bfrag(const float* row, int kb) {
    v8u r;
#pragma unroll
    for (int j = 0; j < 8; ++j) {
        float2 f = *(const float2*)(row + (kb >> 1) + 2 * j + (j >= 4 ? 8 : 0));
        r[j] = (unsigned)f2bf(f.x) | ((unsigned)f2bf(f.y) << 16);
    }
    return __builtin_bit_cast(v16bf, r);
}

__global__ __launch_bounds__(256) void charrnn_wmma_kernel(
    const int*   __restrict__ x,       // (B, T)
    const float* __restrict__ h0,      // (B, H)
    const float* __restrict__ emb,     // (V, E)
    const float* __restrict__ W_cell,  // (H, E+H)
    const float* __restrict__ b_cell,  // (H)
    const float* __restrict__ W_dec,   // (V, H)
    const float* __restrict__ b_dec,   // (V)
    float*       __restrict__ out)     // (B,T,V) logits ++ (B,H) h_final
{
    __shared__ __align__(16) unsigned short h_s[2][MROWS * HIDDEN]; // bf16 ping-pong
    __shared__ __align__(16) unsigned short emb_s[VOCAB * EMBED];   // bf16 table
    __shared__ int x_s[MROWS * TCH];

    const int tid  = threadIdx.x;
    const int wid  = tid >> 5;
    const int lane = tid & 31;
    const int nl   = lane & 15;   // N (and A-matrix M) index within tile
    const int hl   = lane >> 4;   // K-half select
    const int b0   = blockIdx.x * MROWS;

    // ---- one-time staging ----
    for (int i = tid; i < VOCAB * EMBED; i += 256) emb_s[i] = f2bf(emb[i]);
    for (int i = tid; i < MROWS * HIDDEN; i += 256) {
        int m = i >> 7, c = i & (HIDDEN - 1);
        h_s[0][i] = f2bf(h0[(size_t)(b0 + m) * HIDDEN + c]);
    }

    // ---- per-wave weight fragments, resident in VGPRs for all 2048 steps ----
    const int n  = wid * 16 + nl;   // this wave's hidden column (0..127)
    const int kb = hl * 16;
    const float* wrow = W_cell + (size_t)n * (EMBED + HIDDEN);
    v16bf fWe = load_w_bfrag(wrow, kb);                  // We^T tile (K=EMBED)
    v16bf fWh[4];
#pragma unroll
    for (int kc = 0; kc < 4; ++kc)                        // Wh^T tiles (K=128)
        fWh[kc] = load_w_bfrag(wrow + EMBED + kc * 32, kb);
    const float bcell = b_cell[n];

    v16bf fWd[4];
    float bdec = 0.0f;
    if (wid < 6) {                                        // decoder tiles (N<96)
        const float* drow = W_dec + (size_t)n * HIDDEN;
#pragma unroll
        for (int kc = 0; kc < 4; ++kc)
            fWd[kc] = load_w_bfrag(drow + kc * 32, kb);
        bdec = b_dec[n];
    } else {
        v8u z;
#pragma unroll
        for (int j = 0; j < 8; ++j) z[j] = 0u;
#pragma unroll
        for (int kc = 0; kc < 4; ++kc) fWd[kc] = __builtin_bit_cast(v16bf, z);
    }
    __syncthreads();

    float th[8];  // this wave's f32 tanh outputs (also final h)

#pragma unroll 1
    for (int tb = 0; tb < TSTEPS; tb += TCH) {
        // stage token indices for this chunk (coalesced along t)
        for (int i = tid; i < MROWS * TCH; i += 256) {
            int m = i >> 5, ts = i & (TCH - 1);
            x_s[i] = x[(size_t)(b0 + m) * TSTEPS + tb + ts];
        }
        __syncthreads();

#pragma unroll 1
        for (int ts = 0; ts < TCH; ++ts) {
            const int t  = tb + ts;
            const int rd = t & 1, wr = rd ^ 1;
            const unsigned short* hprev = h_s[rd];

            // A fragment: embedded tokens e_t (16x32), row m = nl
            const int idx = x_s[nl * TCH + ts];
            v16bf aE = load_lds_afrag(emb_s, idx * 4 + hl);

            v8f acc;
#pragma unroll
            for (int j = 0; j < 8; ++j) acc[j] = bcell;   // bias broadcast

            acc = __builtin_amdgcn_wmma_f32_16x16x32_bf16(
                    false, aE, false, fWe, (short)0, acc, false, false);
#pragma unroll
            for (int kc = 0; kc < 4; ++kc) {
                v16bf aH = load_lds_afrag(hprev, nl * 16 + kc * 4 + hl);
                acc = __builtin_amdgcn_wmma_f32_16x16x32_bf16(
                        false, aH, false, fWh[kc], (short)0, acc, false, false);
            }

            // hardware tanh (TRANS32) — one op per element, no divergence
#pragma unroll
            for (int j = 0; j < 8; ++j) th[j] = htanh(acc[j]);

            // publish h_next (bf16) — C/D layout: m = hl*8 + j, col = n
            unsigned short* hnext = h_s[wr];
#pragma unroll
            for (int j = 0; j < 8; ++j)
                hnext[(hl * 8 + j) * HIDDEN + n] = f2bf(th[j]);

            __syncthreads();   // single barrier/step thanks to ping-pong

            // decoder: logits = h_next @ W_dec^T + b_dec (waves 0..5)
            if (wid < 6) {
                v8f acc2;
#pragma unroll
                for (int j = 0; j < 8; ++j) acc2[j] = bdec;
#pragma unroll
                for (int kc = 0; kc < 4; ++kc) {
                    v16bf aH = load_lds_afrag(hnext, nl * 16 + kc * 4 + hl);
                    acc2 = __builtin_amdgcn_wmma_f32_16x16x32_bf16(
                            false, aH, false, fWd[kc], (short)0, acc2, false, false);
                }
                // streaming output: non-temporal stores, don't pollute caches
#pragma unroll
                for (int j = 0; j < 8; ++j) {
                    int m = hl * 8 + j;
                    __builtin_nontemporal_store(
                        acc2[j],
                        &out[((size_t)(b0 + m) * TSTEPS + t) * VOCAB + n]);
                }
            }
        }
    }

    // final hidden state (f32, pre-bf16-rounding), appended after logits
    float* hfin = out + (size_t)BATCH * TSTEPS * VOCAB;
#pragma unroll
    for (int j = 0; j < 8; ++j) {
        int m = hl * 8 + j;
        hfin[(size_t)(b0 + m) * HIDDEN + n] = th[j];
    }
}

extern "C" void kernel_launch(void* const* d_in, const int* in_sizes, int n_in,
                              void* d_out, int out_size, void* d_ws, size_t ws_size,
                              hipStream_t stream) {
    const int*   x      = (const int*)  d_in[0];
    const float* h0     = (const float*)d_in[1];
    const float* emb    = (const float*)d_in[2];
    const float* W_cell = (const float*)d_in[3];
    const float* b_cell = (const float*)d_in[4];
    const float* W_dec  = (const float*)d_in[5];
    const float* b_dec  = (const float*)d_in[6];
    float* out = (float*)d_out;
    (void)in_sizes; (void)n_in; (void)out_size; (void)d_ws; (void)ws_size;

    dim3 grid(BATCH / MROWS);   // 32 workgroups
    dim3 block(256);            // 8 wave32s: one 16-wide N-slice each
    charrnn_wmma_kernel<<<grid, block, 0, stream>>>(x, h0, emb, W_cell, b_cell,
                                                    W_dec, b_dec, out);
}
